// dragonet_model_nam_22204980920729
// MI455X (gfx1250) — matrix operand order repaired and checked by reading, not compiled, not run on hardware
//
#include <hip/hip_runtime.h>
#include <hip/hip_bf16.h>

typedef __attribute__((ext_vector_type(16))) __bf16 v16bf;
typedef __attribute__((ext_vector_type(8)))  float  v8f;

constexpr int Bsz  = 4096;   // batch
constexpr int Fsz  = 100;    // features
constexpr int Hsz  = 128;    // hidden
constexpr int H2sz = 64;     // head hidden
constexpr int Tsz  = 2;      // treatments
constexpr int BT   = 128;    // batch tile per workgroup
constexpr int KP   = 136;    // padded LDS row stride (elements, 272B rows, 16B aligned)
constexpr int FG   = 10;     // feature groups (partial-sum slots in workspace)
constexpr int FPG  = Fsz / FG; // features per group
constexpr int WGS  = 256;    // 8 waves of 32

union Frag  { v16bf v; uint4 q[2]; };
union Pack4 { __bf16 e[4]; uint2 u; };
union Pack8 { __bf16 e[8]; uint4 q; };
static_assert(sizeof(Frag) == 32 && sizeof(Pack4) == 8 && sizeof(Pack8) == 16, "sizes");

__device__ __forceinline__ float elu1(float v) {
    return v > 0.f ? v : (__expf(v) - 1.f);
}

// Full 128x128x128 GEMM strip: this wave's 16 M-rows, all 8 N-tiles, 4 K-steps.
// B fragments double-buffered (one tile lookahead) so wmma can wait at dscnt<=2.
__device__ __forceinline__ void gemm128(const __bf16* __restrict__ hin,
                                        const __bf16* __restrict__ w,
                                        v8f acc[8], int mrow, int lm, int ksel, int bk)
{
    const v8f vzero = {0.f,0.f,0.f,0.f,0.f,0.f,0.f,0.f};
    #pragma unroll
    for (int nt = 0; nt < 8; ++nt) acc[nt] = vzero;
    #pragma unroll
    for (int ks = 0; ks < 4; ++ks) {
        const int kb = ks * 32;
        Frag A;
        const __bf16* ap = hin + (size_t)mrow * KP + kb + ksel;
        A.q[0] = *(const uint4*)(ap);
        A.q[1] = *(const uint4*)(ap + 16);
        Frag Bf[2];
        const __bf16* bp0 = w + (size_t)lm * KP + kb + bk;
        Bf[0].q[0] = *(const uint4*)(bp0);
        Bf[0].q[1] = *(const uint4*)(bp0 + 16);
        #pragma unroll
        for (int nt = 0; nt < 8; ++nt) {
            if (nt < 7) {
                const __bf16* bp = w + (size_t)((nt + 1) * 16 + lm) * KP + kb + bk;
                Bf[(nt + 1) & 1].q[0] = *(const uint4*)(bp);
                Bf[(nt + 1) & 1].q[1] = *(const uint4*)(bp + 16);
            }
            acc[nt] = __builtin_amdgcn_wmma_f32_16x16x32_bf16(
                false, A.v, false, Bf[nt & 1].v, (short)0, acc[nt], false, false);
        }
    }
}

// bias + optional ELU, write bf16 activations to LDS; optional register running sum.
__device__ __forceinline__ void epilogue(v8f acc[8], const float* __restrict__ bias,
                                         __bf16* __restrict__ hout,
                                         int moff, int lm, bool do_elu, v8f* bbsum)
{
    #pragma unroll
    for (int nt = 0; nt < 8; ++nt) {
        const int n = nt * 16 + lm;
        const float bv = bias[n];
        #pragma unroll
        for (int r = 0; r < 8; ++r) {
            float v = acc[nt][r] + bv;
            if (do_elu) v = elu1(v);
            if (bbsum) bbsum[nt][r] += v;
            hout[(size_t)(moff + r) * KP + n] = (__bf16)v;
        }
    }
}

// Stage a 128x128 fp32 weight matrix into LDS transposed [n][k] as bf16, plus bias.
__device__ __forceinline__ void stage_weight(const float* __restrict__ Wsrc,
                                             const float* __restrict__ bsrc,
                                             __bf16* __restrict__ shw,
                                             float* __restrict__ shb, int tid)
{
    #pragma unroll 4
    for (int it = 0; it < (Hsz * Hsz) / (WGS * 4); ++it) {   // 16 iterations
        const int i = (tid + it * WGS) * 4;
        const int k = i >> 7, n = i & (Hsz - 1);
        const float4 w4 = *(const float4*)(Wsrc + (size_t)k * Hsz + n);
        shw[(n + 0) * KP + k] = (__bf16)w4.x;
        shw[(n + 1) * KP + k] = (__bf16)w4.y;
        shw[(n + 2) * KP + k] = (__bf16)w4.z;
        shw[(n + 3) * KP + k] = (__bf16)w4.w;
    }
    if (tid < Hsz) shb[tid] = bsrc[tid];
}

// Collapse the two head linears (no nonlinearity between them) into w_eff[F,H,T], c_eff[F,T].
__global__ __launch_bounds__(WGS) void nam_head_collapse(
    const float* __restrict__ Wh1, const float* __restrict__ bh1,
    const float* __restrict__ Wh2, const float* __restrict__ bh2,
    float* __restrict__ weff, float* __restrict__ ceff)
{
    const int f = blockIdx.x;
    const int tid = threadIdx.x;
    const int t = tid >> 7;          // 0..1
    const int h = tid & (Hsz - 1);   // 0..127
    const float* w1p = Wh1 + (((size_t)f * Tsz + t) * Hsz + h) * H2sz;
    const float* w2p = Wh2 + ((size_t)f * Tsz + t) * H2sz;
    float s = 0.f;
    #pragma unroll 8
    for (int m = 0; m < H2sz; ++m) s += w1p[m] * w2p[m];
    weff[((size_t)f * Hsz + h) * Tsz + t] = s;
    if (h == 0) {
        const float* b1p = bh1 + ((size_t)f * Tsz + t) * H2sz;
        float c = bh2[(size_t)f * Tsz + t];
        for (int m = 0; m < H2sz; ++m) c += b1p[m] * w2p[m];
        ceff[(size_t)f * Tsz + t] = c;
    }
}

__global__ __launch_bounds__(WGS) void nam_backbone(
    const float* __restrict__ x,  const int* __restrict__ a,
    const float* __restrict__ W1, const float* __restrict__ b1,
    const float* __restrict__ W2, const float* __restrict__ b2,
    const float* __restrict__ W3, const float* __restrict__ b3,
    const float* __restrict__ W4, const float* __restrict__ b4,
    const float* __restrict__ weff, const float* __restrict__ ceff,
    float* __restrict__ ws_bb, float* __restrict__ ws_ys)
{
    __shared__ __align__(16) __bf16 sh_w[Hsz * KP];       // transposed weights [n][k], bf16
    __shared__ __align__(16) __bf16 sh_h[2][BT * KP];     // ping-pong activations [b][k], bf16
    __shared__ float sh_bias[Hsz];
    __shared__ float sh_x[BT];

    const int tile_b0 = blockIdx.x * BT;
    const int g  = blockIdx.y;        // feature group
    const int f0 = g * FPG;

    const int tid  = threadIdx.x;
    const int wave = tid >> 5;
    const int lane = tid & 31;
    const int lm   = lane & 15;
    const int ksel = (lane < 16) ? 0 : 8;    // A-frag K sub-select
    const int bk   = (lane < 16) ? 0 : 8;    // B-frag K sub-select
    const int mbase = wave * 16;
    const int mrow  = mbase + lm;
    const int moff  = mbase + ((lane < 16) ? 0 : 8);

    const v8f vzero = {0.f,0.f,0.f,0.f,0.f,0.f,0.f,0.f};
    v8f bbsum[8];
    #pragma unroll
    for (int nt = 0; nt < 8; ++nt) bbsum[nt] = vzero;

    float y_acc = 0.f;
    int my_t = 0;
    if (tid < BT) my_t = a[tile_b0 + tid];

    for (int fi = 0; fi < FPG; ++fi) {
        const int f = f0 + fi;
        __syncthreads();
        if (tid < BT) sh_x[tid] = x[(size_t)(tile_b0 + tid) * Fsz + f];
        __syncthreads();

        // ---- layer 1 (elementwise ELU) -> sh_h[0]; stage W2^T + b2 ----
        const float* W1f = W1 + (size_t)f * Hsz;
        const float* b1f = b1 + (size_t)f * Hsz;
        #pragma unroll 4
        for (int it = 0; it < (BT * Hsz) / (WGS * 4); ++it) {  // 16 iterations
            const int i = (tid + it * WGS) * 4;
            const int b = i >> 7, k = i & (Hsz - 1);
            const float xv = sh_x[b];
            const float4 w4 = *(const float4*)(W1f + k);
            const float4 c4 = *(const float4*)(b1f + k);
            Pack4 p;
            p.e[0] = (__bf16)elu1(xv * w4.x + c4.x);
            p.e[1] = (__bf16)elu1(xv * w4.y + c4.y);
            p.e[2] = (__bf16)elu1(xv * w4.z + c4.z);
            p.e[3] = (__bf16)elu1(xv * w4.w + c4.w);
            *(uint2*)(&sh_h[0][b * KP + k]) = p.u;
        }
        stage_weight(W2 + (size_t)f * Hsz * Hsz, b2 + (size_t)f * Hsz, sh_w, sh_bias, tid);
        __syncthreads();

        v8f acc[8];

        // ---- layer 2: GEMM + ELU -> sh_h[1] ----
        gemm128(sh_h[0], sh_w, acc, mrow, lm, ksel, bk);
        epilogue(acc, sh_bias, sh_h[1], moff, lm, true, (v8f*)nullptr);
        __syncthreads();

        stage_weight(W3 + (size_t)f * Hsz * Hsz, b3 + (size_t)f * Hsz, sh_w, sh_bias, tid);
        __syncthreads();

        // ---- layer 3: GEMM + ELU -> sh_h[0] ----
        gemm128(sh_h[1], sh_w, acc, mrow, lm, ksel, bk);
        epilogue(acc, sh_bias, sh_h[0], moff, lm, true, (v8f*)nullptr);
        __syncthreads();

        stage_weight(W4 + (size_t)f * Hsz * Hsz, b4 + (size_t)f * Hsz, sh_w, sh_bias, tid);
        __syncthreads();

        // ---- layer 4: GEMM (no ELU) -> bb in sh_h[1]; running backbone sum in regs ----
        gemm128(sh_h[0], sh_w, acc, mrow, lm, ksel, bk);
        epilogue(acc, sh_bias, sh_h[1], moff, lm, false, bbsum);
        __syncthreads();

        // ---- collapsed head: y[b] += bb[b,:] . w_eff[f,:,a[b]] + c_eff[f,a[b]] ----
        if (tid < BT) {
            const uint4* rowq = (const uint4*)&sh_h[1][tid * KP];
            const float2* we2 = (const float2*)(weff + (size_t)f * Hsz * Tsz);
            float s = 0.f;
            #pragma unroll 4
            for (int hh = 0; hh < Hsz / 8; ++hh) {   // 16 iterations, 8 bf16 per load
                Pack8 p; p.q = rowq[hh];
                #pragma unroll
                for (int j = 0; j < 8; ++j) {
                    const float2 w = we2[hh * 8 + j];
                    s += (float)p.e[j] * (my_t ? w.y : w.x);
                }
            }
            y_acc += s + ceff[(size_t)f * Tsz + my_t];
        }
    }

    // one-time write of this group's partials (exclusive ownership -> deterministic)
    float* gptr = ws_bb + ((size_t)g * Bsz + tile_b0 + moff) * Hsz + lm;
    #pragma unroll
    for (int nt = 0; nt < 8; ++nt) {
        #pragma unroll
        for (int r = 0; r < 8; ++r) {
            gptr[(size_t)r * Hsz + nt * 16] = bbsum[nt][r];
        }
    }
    if (tid < BT) ws_ys[(size_t)g * Bsz + tile_b0 + tid] = y_acc;
}

__global__ __launch_bounds__(WGS) void nam_reduce(const float* __restrict__ ws_bb,
                                                  const float* __restrict__ ws_ys,
                                                  float* __restrict__ out)
{
    const size_t NBH = (size_t)Bsz * Hsz;
    const size_t i = (size_t)blockIdx.x * WGS + threadIdx.x;
    if (i < NBH) {
        float s = 0.f;
        #pragma unroll
        for (int gg = 0; gg < FG; ++gg) s += ws_bb[(size_t)gg * NBH + i];
        out[i] = s;
    }
    if (i < (size_t)Bsz) {
        float s = 0.f;
        #pragma unroll
        for (int gg = 0; gg < FG; ++gg) s += ws_ys[(size_t)gg * Bsz + i];
        out[NBH + i] = s;
    }
}

extern "C" void kernel_launch(void* const* d_in, const int* in_sizes, int n_in,
                              void* d_out, int out_size, void* d_ws, size_t ws_size,
                              hipStream_t stream)
{
    (void)in_sizes; (void)n_in; (void)out_size; (void)ws_size;
    const float* x   = (const float*)d_in[0];
    const int*   a   = (const int*)  d_in[1];
    const float* W1  = (const float*)d_in[2];
    const float* b1  = (const float*)d_in[3];
    const float* W2  = (const float*)d_in[4];
    const float* b2  = (const float*)d_in[5];
    const float* W3  = (const float*)d_in[6];
    const float* b3  = (const float*)d_in[7];
    const float* W4  = (const float*)d_in[8];
    const float* b4  = (const float*)d_in[9];
    const float* Wh1 = (const float*)d_in[10];
    const float* bh1 = (const float*)d_in[11];
    const float* Wh2 = (const float*)d_in[12];
    const float* bh2 = (const float*)d_in[13];
    float* out = (float*)d_out;

    // workspace layout (all fp32): partial backbone sums, partial y sums, collapsed head
    float* ws_bb = (float*)d_ws;                       // FG * B * H
    float* ws_ys = ws_bb + (size_t)FG * Bsz * Hsz;     // FG * B
    float* weff  = ws_ys + (size_t)FG * Bsz;           // F * H * T
    float* ceff  = weff  + (size_t)Fsz * Hsz * Tsz;    // F * T

    nam_head_collapse<<<dim3(Fsz), dim3(WGS), 0, stream>>>(Wh1, bh1, Wh2, bh2, weff, ceff);
    nam_backbone<<<dim3(Bsz / BT, FG), dim3(WGS), 0, stream>>>(
        x, a, W1, b1, W2, b2, W3, b3, W4, b4, weff, ceff, ws_bb, ws_ys);
    nam_reduce<<<dim3((Bsz * Hsz + WGS - 1) / WGS), dim3(WGS), 0, stream>>>(ws_bb, ws_ys, out);
}
